// _VQLayer_67224828117166
// MI455X (gfx1250) — matrix operand order repaired and checked by reading, not compiled, not run on hardware
//
#include <hip/hip_runtime.h>
#include <hip/hip_bf16.h>

typedef __attribute__((ext_vector_type(16))) _Float16 v16h;
typedef __attribute__((ext_vector_type(8)))  _Float16 h8;
typedef __attribute__((ext_vector_type(8)))  float    v8f;

#define Bn 16
#define Cn 64
#define HWn 4096           // 64*64
#define NTOK 65536         // Bn*HWn
#define NCODES 4096

// ---------------------------------------------------------------------------
// prep_z: (B,C,H,W) f32 -> token-major f16 zt[NTOK][64], LDS tile transpose.
// grid: Bn * (HWn/256) = 256 blocks, 256 threads.
// ---------------------------------------------------------------------------
__global__ void prep_z_kernel(const float* __restrict__ z, _Float16* __restrict__ zt) {
    __shared__ _Float16 tile[16][258];   // padded to avoid bank conflicts
    const int t      = threadIdx.x;
    const int b      = blockIdx.x >> 4;
    const int hwbase = (blockIdx.x & 15) * 256;
    const int cl = t & 15, tg = t >> 4;
    for (int cc = 0; cc < 64; cc += 16) {
        __syncthreads();
        #pragma unroll
        for (int i = 0; i < 16; ++i) {
            const int c = cc + i;
            tile[i][t] = (_Float16)z[(((size_t)(b * Cn + c)) << 12) + hwbase + t];
        }
        __syncthreads();
        #pragma unroll
        for (int p = 0; p < 16; ++p) {
            const int tokl = p * 16 + tg;
            zt[((size_t)(b * HWn + hwbase + tokl)) * 64 + cc + cl] = tile[cl][tokl];
        }
    }
}

// ---------------------------------------------------------------------------
// prep_emb: emb f32 -> f16 ebf[4096][64] + squared-norm table.
// ---------------------------------------------------------------------------
__global__ void prep_emb_kernel(const float* __restrict__ emb,
                                _Float16* __restrict__ ebf,
                                float* __restrict__ enorm) {
    const int n = blockIdx.x * blockDim.x + threadIdx.x;
    const float* er = emb + (size_t)n * Cn;
    float s = 0.0f;
    #pragma unroll 8
    for (int c = 0; c < Cn; ++c) {
        const float v = er[c];
        s += v * v;
        ebf[(size_t)n * Cn + c] = (_Float16)v;
    }
    enorm[n] = s;
}

// ---------------------------------------------------------------------------
// CDNA5 async copy: 16B global -> LDS, tracked by ASYNCcnt.
// LDS destination offset is derived from the real shared-memory pointer
// (generic LDS pointer = {aperture, wave-relative offset[31:0]}, ISA 10.2);
// the ptrtoint also CAPTURES the buffer so consume-side LDS loads are real.
// ---------------------------------------------------------------------------
__device__ __forceinline__ void async_ld16(const _Float16* lds_ptr, const char* g) {
    const unsigned off = (unsigned)(size_t)(const void*)lds_ptr;
    asm volatile("global_load_async_to_lds_b128 %0, %1, off"
                 :: "v"(off), "v"(g) : "memory");
}

// ---------------------------------------------------------------------------
// vq_argmin: fused GEMM + argmin.
//  - one wave handles 2 token tiles (32 tokens) vs all 4096 codes
//  - the 8 waves of a block share codebook tiles staged in LDS:
//    64 codes (8KB) per stage, double-buffered, async global->LDS copies
//  - per stage per wave: 4 code tiles * 2 token tiles * 2 WMMAs = 16 WMMAs
//  - stage loop kept rolled (unroll(disable)) to bound VGPR liveness and
//    avoid the scratch spills seen when two stages were overlapped
// grid: 256 blocks * 256 threads  (2048 waves * 32 tokens = 65536 tokens)
// ---------------------------------------------------------------------------
#define G_TILES   4                      // code tiles (of 16) per stage
#define STAGE_CODES (G_TILES * 16)       // 64 codes
#define STAGE_HALF  (STAGE_CODES * 64)   // 4096 f16 elements = 8192 bytes
#define NSTAGES   (NCODES / STAGE_CODES) // 64

__global__ void __launch_bounds__(256)
vq_argmin_kernel(const _Float16* __restrict__ zt,
                 const _Float16* __restrict__ ebf,
                 const float* __restrict__ enorm,
                 int* __restrict__ idx_out) {
    __shared__ _Float16 buf[2][STAGE_HALF];   // 16 KB double buffer

    const int tid  = threadIdx.x;
    const int lane = tid & 31;
    const int hf   = lane >> 4;      // lane half
    const int nlo  = lane & 15;
    const int wave = blockIdx.x * (blockDim.x >> 5) + (tid >> 5);
    const int tokbase = wave * 32;   // 2 token tiles per wave

    // ---- A fragments (16x32 f16, k = 16*hf + s per lane half), 2 token tiles resident
    v16h a0[2], a1[2];
    #pragma unroll
    for (int tt = 0; tt < 2; ++tt) {
        const _Float16* za = zt + (size_t)(tokbase + tt * 16 + nlo) * 64;
        *(h8*)&a0[tt]       = *(const h8*)(za + 0  + hf * 16);
        *((h8*)&a0[tt] + 1) = *(const h8*)(za + 8  + hf * 16);
        *(h8*)&a1[tt]       = *(const h8*)(za + 32 + hf * 16);
        *((h8*)&a1[tt] + 1) = *(const h8*)(za + 40 + hf * 16);
    }

    float minv[2][8];
    int   mini[2][8];
    #pragma unroll
    for (int tt = 0; tt < 2; ++tt)
        #pragma unroll
        for (int j = 0; j < 8; ++j) { minv[tt][j] = 3.4028235e38f; mini[tt][j] = 0; }

    const char* gsrc = (const char*)ebf;

    // prologue: issue async copies for stage 0 (2 x 16B chunks per thread)
    async_ld16(&buf[0][tid * 8],        gsrc + (size_t)tid * 16);
    async_ld16(&buf[0][tid * 8 + 2048], gsrc + (size_t)tid * 16 + 4096);

    #pragma clang loop unroll(disable)
    for (int s = 0; s < NSTAGES; ++s) {
        const int bs = s & 1;
        if (s + 1 < NSTAGES) {
            // issue next stage into the other buffer (safe: barrier at end of
            // iteration s-1 guaranteed all waves finished reading it)
            const char* src = gsrc + (size_t)(s + 1) * 8192;
            async_ld16(&buf[bs ^ 1][tid * 8],        src + (size_t)tid * 16);
            async_ld16(&buf[bs ^ 1][tid * 8 + 2048], src + (size_t)tid * 16 + 4096);
            // async loads retire in order: <=2 outstanding => stage s complete
            asm volatile("s_wait_asynccnt 0x2" ::: "memory");
        } else {
            asm volatile("s_wait_asynccnt 0x0" ::: "memory");
        }
        __syncthreads();   // all waves' stage-s data visible in LDS

        const int stagebase = s * STAGE_CODES;
        #pragma unroll
        for (int g = 0; g < G_TILES; ++g) {
            const int code = stagebase + g * 16 + nlo;
            // B fragment: 16 contiguous K halves of code row, from LDS
            const _Float16* eb = &buf[bs][(g * 16 + nlo) * 64 + hf * 16];
            const v16h b0 = *(const v16h*)(eb);
            const v16h b1 = *(const v16h*)(eb + 32);
            const float en = enorm[code];

            #pragma unroll
            for (int tt = 0; tt < 2; ++tt) {
                v8f c = {};
                c = __builtin_amdgcn_wmma_f32_16x16x32_f16(false, a0[tt], false, b0,
                                                           (short)0, c, false, false);
                c = __builtin_amdgcn_wmma_f32_16x16x32_f16(false, a1[tt], false, b1,
                                                           (short)0, c, false, false);
                #pragma unroll
                for (int j = 0; j < 8; ++j) {
                    const float d = en - 2.0f * c[j];  // ||e||^2 - 2 z.e
                    if (d < minv[tt][j]) { minv[tt][j] = d; mini[tt][j] = code; }
                }
            }
        }
        __syncthreads();   // all waves done reading stage s buffer
    }

    // ---- cross-lane argmin reduce (16-lane groups), tie-break smaller index
    #pragma unroll
    for (int tt = 0; tt < 2; ++tt) {
        #pragma unroll
        for (int j = 0; j < 8; ++j) {
            float v = minv[tt][j];
            int   ix = mini[tt][j];
            #pragma unroll
            for (int m = 8; m >= 1; m >>= 1) {
                const float ov = __shfl_xor(v, m, 32);
                const int   oi = __shfl_xor(ix, m, 32);
                if (ov < v || (ov == v && oi < ix)) { v = ov; ix = oi; }
            }
            if (nlo == 0) {
                const int mrow = j + (hf << 3);
                idx_out[tokbase + tt * 16 + mrow] = ix;
            }
        }
    }
}

// ---------------------------------------------------------------------------
// gather: out[b][c][hw] = emb[idx[b*HW+hw]][c]
// ---------------------------------------------------------------------------
__global__ void gather_kernel(const float* __restrict__ emb,
                              const int* __restrict__ idx,
                              float* __restrict__ out) {
    const int gid = blockIdx.x * 256 + threadIdx.x;   // over B*C*HW = 4,194,304
    const int hw  = gid & (HWn - 1);
    const int bc  = gid >> 12;
    const int c   = bc & (Cn - 1);
    const int b   = bc >> 6;
    const int token = (b << 12) + hw;
    out[gid] = emb[(size_t)idx[token] * Cn + c];
}

// ---------------------------------------------------------------------------
extern "C" void kernel_launch(void* const* d_in, const int* in_sizes, int n_in,
                              void* d_out, int out_size, void* d_ws, size_t ws_size,
                              hipStream_t stream) {
    const float* z   = (const float*)d_in[0];   // (16,64,64,64) f32
    const float* emb = (const float*)d_in[1];   // (4096,64)     f32
    float* out = (float*)d_out;                 // (16,64,64,64) f32

    char* ws = (char*)d_ws;
    _Float16* zt    = (_Float16*)(ws);                 // 8388608 B
    _Float16* ebf   = (_Float16*)(ws + 8388608);       // 524288 B
    float*    enorm = (float*)   (ws + 8912896);       // 16384 B
    int*      idx   = (int*)     (ws + 8929280);       // 262144 B

    prep_z_kernel   <<<256,   256, 0, stream>>>(z, zt);
    prep_emb_kernel <<<16,    256, 0, stream>>>(emb, ebf, enorm);
    vq_argmin_kernel<<<256,   256, 0, stream>>>(zt, ebf, enorm, idx);
    gather_kernel   <<<16384, 256, 0, stream>>>(emb, idx, out);
}
